// VaLPACa_Net_77240691851804
// MI455X (gfx1250) — matrix operand order repaired and checked by reading, not compiled, not run on hardware
//
#include <hip/hip_runtime.h>
#include <hip/hip_bf16.h>
#include <cstdint>
#include <cstddef>

// ---------------------------------------------------------------------------
// VaLPACa forward for MI455X (gfx1250, wave32, WMMA).
// All heavy math runs through v_wmma_f32_16x16x32_bf16.
//  - Time-parallel input projections: big bf16 WMMA GEMMs (1x4 tile blocking,
//    A-fragment reused across 4 N-tiles).
//  - Encoder recurrences: persistent kernel, 6 blocks (one per GRU direction),
//    hidden state resident in LDS (bf16, padded stride).
//  - Decoder: single persistent kernel with a grid barrier; 7 WMMA phases/step.
// ---------------------------------------------------------------------------

typedef __bf16 bf16;
typedef bf16  v16bf __attribute__((ext_vector_type(16)));
typedef bf16  v8bf  __attribute__((ext_vector_type(8)));
typedef float v8f   __attribute__((ext_vector_type(8)));

#define DEV __device__ __forceinline__

constexpr int N_ = 512, T_ = 256, B_ = 128;
constexpr int GL = 64, UL = 16, DC = 128;
constexpr int GEN = 256, F_ = 64, PAD_ = 3, OC = 256, OL = 128;
constexpr int TP = T_ + 2 * PAD_;            // 262 padded steps for obs encoder
constexpr float CLIPV = 5.0f;
constexpr float DECAY = 1.0f - 1.0f / 10.0f; // 1 - 1/TAU
constexpr int HSTR = 272;                    // LDS row stride (256 + pad), 16B aligned

DEV v8f vz() { v8f z = {0.f,0.f,0.f,0.f,0.f,0.f,0.f,0.f}; return z; }
DEV float sigmoidf_(float x) { return 1.f / (1.f + __expf(-x)); }

// A fragment: 16x32 bf16, row-major source. ISA 7.12.2 layout:
// lane<16: row m0+l, elems 0..7 = K k0..k0+7, elems 8..15 = K k0+16..23
// lane>=16: row m0+(l-16), K k0+8..15 and k0+24..31.
DEV v16bf load_a_frag(const bf16* base, int ld, int m0, int k0, int lane) {
  int l = lane & 15, hi = lane >> 4;
  const bf16* p = base + (size_t)(m0 + l) * ld + k0 + hi * 8;
  v8bf lo  = *(const v8bf*)p;
  v8bf hi8 = *(const v8bf*)(p + 16);
  return __builtin_shufflevector(lo, hi8, 0,1,2,3,4,5,6,7,8,9,10,11,12,13,14,15);
}

// B fragment: 32x16 where B[k,n] = W[n,k] (W row-major, ldw = K).
// lane<16: col n0+l, K k0..k0+7 and k0+16..23 ; lane>=16: col n0+(l-16), K k0+8..15 and k0+24..31.
DEV v16bf load_b_frag(const bf16* Wt, int ldw, int n0, int k0, int lane) {
  int l = lane & 15, hi = lane >> 4;
  const bf16* p = Wt + (size_t)(n0 + l) * ldw + k0 + hi * 8;
  v8bf lo  = *(const v8bf*)p;
  v8bf hi8 = *(const v8bf*)(p + 16);
  return __builtin_shufflevector(lo, hi8, 0,1,2,3,4,5,6,7,8,9,10,11,12,13,14,15);
}

DEV v8f wmma_bf(v16bf a, v16bf b, v8f c) {
  return __builtin_amdgcn_wmma_f32_16x16x32_bf16(false, a, false, b, (short)0, c,
                                                 false, false);
}

// ---------------------------------------------------------------------------
// Small utility kernels
// ---------------------------------------------------------------------------
__global__ void cast_rows_kernel(bf16* dst, const float* src, int rows, int scols,
                                 int dstride) {
  int i = blockIdx.x * blockDim.x + threadIdx.x;
  int total = rows * dstride;
  if (i >= total) return;
  int r = i / dstride, c = i % dstride;
  dst[i] = (c < scols) ? (bf16)src[(size_t)r * scols + c] : (bf16)0.f;
}

// x (B,T,N) f32 -> x_pad (TP,B,N) bf16 at time offset PAD (pad rows pre-zeroed)
__global__ void transpose_cast_x_kernel(bf16* dst, const float* x) {
  size_t i = (size_t)blockIdx.x * blockDim.x + threadIdx.x;
  size_t total = (size_t)T_ * B_ * N_;
  if (i >= total) return;
  int n = (int)(i % N_);
  size_t r = i / N_;
  int b = (int)(r % B_);
  int t = (int)(r / B_);
  dst[(((size_t)(t + PAD_)) * B_ + b) * N_ + n] = (bf16)x[((size_t)b * T_ + t) * N_ + n];
}

__global__ void lin_naive_kernel(float* outf, bf16* outbf, const float* A,
                                 const float* Wt, const float* b, int M, int N, int K) {
  int i = blockIdx.x * blockDim.x + threadIdx.x;
  if (i >= M * N) return;
  int m = i / N, n = i % N;
  float s = b ? b[n] : 0.f;
  const float* a = A + (size_t)m * K;
  const float* w = Wt + (size_t)n * K;
  for (int k = 0; k < K; ++k) s += a[k] * w[k];
  if (outf) outf[i] = s;
  if (outbf) outbf[i] = (bf16)s;
}

__global__ void sample_g0_kernel(float* g0, const float* gm, const float* glv,
                                 const float* eps, int n) {
  int i = blockIdx.x * blockDim.x + threadIdx.x;
  if (i < n) g0[i] = gm[i] + eps[i] * __expf(0.5f * glv[i]);
}

// ---------------------------------------------------------------------------
// Generic WMMA GEMM: C[M,N] = A[M,K](bf16) @ W[N,K]^T (bf16) + bias
// Output: bf16 and/or f32 (optionally exp()). Requires N % 64 == 0.
// Each wave computes one M-tile x four N-tiles so every A fragment is reused
// across 4 WMMAs (fragment loads per WMMA: 2.5 b128 instead of 4).
// ---------------------------------------------------------------------------
__global__ void __launch_bounds__(256)
gemm_bf16_kernel(const bf16* A, int lda, const bf16* Wt, int ldw, const float* bias,
                 bf16* outbf, float* outf, int M, int N, int K, int do_exp) {
  int lane = threadIdx.x & 31;
  int wave = (int)((blockIdx.x * blockDim.x + threadIdx.x) >> 5);
  int nwaves = (int)((gridDim.x * blockDim.x) >> 5);
  int mt_n = M >> 4, ng_n = N >> 6;            // N-tile groups of 4
  for (int grp = wave; grp < mt_n * ng_n; grp += nwaves) {
    int mt = grp / ng_n, ng = grp % ng_n;
    int m0 = mt * 16, nb0 = ng * 64;
    v8f acc[4] = {vz(), vz(), vz(), vz()};
    for (int k0 = 0; k0 < K; k0 += 32) {
      v16bf a = load_a_frag(A, lda, m0, k0, lane);
#pragma unroll
      for (int j = 0; j < 4; ++j)
        acc[j] = wmma_bf(a, load_b_frag(Wt, ldw, nb0 + j * 16, k0, lane), acc[j]);
    }
    int li = lane & 15, hb = lane >> 4;
#pragma unroll
    for (int j = 0; j < 4; ++j) {
      int col = nb0 + j * 16 + li;
      float bv = bias ? bias[col] : 0.f;
#pragma unroll
      for (int i = 0; i < 8; ++i) {
        int row = m0 + i + 8 * hb;
        float v = acc[j][i] + bv;
        if (outbf) outbf[(size_t)row * N + col] = (bf16)v;
        if (outf)  outf[(size_t)row * N + col] = do_exp ? __expf(v) : v;
      }
    }
  }
}

// ---------------------------------------------------------------------------
// Persistent encoder kernel: 6 blocks x 1024 threads (32 waves).
// Each block runs one GRU direction (h=256, B=128) over its whole sequence.
// h lives in LDS (bf16). Per step: gh = h @ Wh^T via WMMA; each wave owns one
// 16-row M tile and the r/z/n sections of 4 N positions, so gate math is pure
// register work on the C fragments.
// ---------------------------------------------------------------------------
struct EncOne {
  const bf16* gi;      // (Tlen, 128, 768) = x@Wi^T + bi  (precomputed)
  const bf16* wh;      // (768, 256) bf16
  const float* bh;     // 768
  bf16* out_seq;       // (Tlen, 128, 512), half selected by dir; or null
  float* out_final;    // (128, 512) f32 final hidden, half by dir; or null
  int Tlen;
  int dir;             // 0 fwd, 1 bwd
};
struct EncArgs { EncOne e[6]; };

__global__ void __launch_bounds__(1024) encoder_kernel(EncArgs args) {
  EncOne a = args.e[blockIdx.x];
  extern __shared__ bf16 hbf[];                 // [128][HSTR]
  for (int i = threadIdx.x; i < B_ * HSTR; i += blockDim.x) hbf[i] = (bf16)0.f;
  __syncthreads();
  int w = threadIdx.x >> 5, lane = threadIdx.x & 31;
  int mt = w & 7, g4 = w >> 3;
  int m0 = mt * 16;
  int li = lane & 15, hb = lane >> 4;
  for (int s = 0; s < a.Tlen; ++s) {
    int t = a.dir ? (a.Tlen - 1 - s) : s;
    const bf16* gi_t = a.gi + (size_t)t * B_ * 768;
    v8f acc[4][3];
#pragma unroll
    for (int p = 0; p < 4; ++p)
#pragma unroll
      for (int gg = 0; gg < 3; ++gg) acc[p][gg] = vz();
    for (int k0 = 0; k0 < 256; k0 += 32) {
      v16bf af = load_a_frag(hbf, HSTR, m0, k0, lane);
#pragma unroll
      for (int p = 0; p < 4; ++p) {
        int n0 = (g4 * 4 + p) * 16;
#pragma unroll
        for (int gg = 0; gg < 3; ++gg)
          acc[p][gg] = wmma_bf(af, load_b_frag(a.wh, 256, gg * 256 + n0, k0, lane),
                               acc[p][gg]);
      }
    }
    __syncthreads();   // all reads of h done before overwrite
#pragma unroll
    for (int p = 0; p < 4; ++p) {
      int n0 = (g4 * 4 + p) * 16, col = n0 + li;
#pragma unroll
      for (int i = 0; i < 8; ++i) {
        int row = m0 + i + 8 * hb;
        size_t gb = (size_t)row * 768;
        float gir  = (float)gi_t[gb + col];
        float giz  = (float)gi_t[gb + col + 256];
        float ginn = (float)gi_t[gb + col + 512];
        float ghr = acc[p][0][i] + a.bh[col];
        float ghz = acc[p][1][i] + a.bh[col + 256];
        float ghn = acc[p][2][i] + a.bh[col + 512];
        float r = sigmoidf_(gir + ghr);
        float z = sigmoidf_(giz + ghz);
        float nn = tanhf(ginn + r * ghn);
        float ho = (float)hbf[(size_t)row * HSTR + col];
        float h = (1.f - z) * nn + z * ho;
        h = fminf(fmaxf(h, -CLIPV), CLIPV);
        hbf[(size_t)row * HSTR + col] = (bf16)h;
        if (a.out_seq)
          a.out_seq[((size_t)t * B_ + row) * 512 + a.dir * 256 + col] = (bf16)h;
        if (a.out_final && s == a.Tlen - 1)
          a.out_final[(size_t)row * 512 + a.dir * 256 + col] = h;
      }
    }
    __syncthreads();
  }
}

// ---------------------------------------------------------------------------
// Decoder: persistent grid-synchronized kernel.
// ---------------------------------------------------------------------------
struct DecArgs {
  unsigned* counter; unsigned nblocks;
  const bf16 *Wid, *Whd, *Wum, *Wulv, *Wig, *Whg, *Wf, *Wio, *Who, *Wom, *Wolv, *Ws;
  const float *bh_d, *b_um, *b_ulv, *bi_g, *bh_g, *bh_o, *b_om, *b_olv, *b_s;
  const bf16 *Gi_c, *Gi_o;
  const float *eps_deep, *eps_obs;
  bf16 *dctrl0, *dctrl1, *gen0, *gen1, *octrl0, *octrl1;
  bf16 *factor_bf, *obs_bf, *u_bf, *gin, *factors_all;
  float *obs_f32;
  float *out_factors, *out_geninputs, *out_spikes, *out_data;
};

DEV void gbar(unsigned* cnt, unsigned nb, unsigned& ph) {
  __syncthreads();
  if (threadIdx.x == 0) {
    ph += 1;
    __threadfence();
    atomicAdd(cnt, 1u);
    volatile unsigned* vc = (volatile unsigned*)cnt;
    unsigned target = nb * ph;
    while (*vc < target) { __builtin_amdgcn_s_sleep(2); }
    __threadfence();
  }
  __syncthreads();
}

DEV void accum_gates(v8f acc[3], const bf16* A, int lda, const bf16* Wt, int ldw,
                     int K, int gstride, int m0, int n0, int lane) {
  for (int k0 = 0; k0 < K; k0 += 32) {
    v16bf a = load_a_frag(A, lda, m0, k0, lane);
    acc[0] = wmma_bf(a, load_b_frag(Wt, ldw, n0,               k0, lane), acc[0]);
    acc[1] = wmma_bf(a, load_b_frag(Wt, ldw, gstride + n0,     k0, lane), acc[1]);
    acc[2] = wmma_bf(a, load_b_frag(Wt, ldw, 2 * gstride + n0, k0, lane), acc[2]);
  }
}

// One GRU update tile: gi from (optional) precomputed buffer + (optional)
// extra-input GEMM + bi ; gh from h_cur @ Wh^T + bh ; writes h_new.
DEV void gru_phase(const bf16* gi_t, const float* bi, int H, const float* bh,
                   const bf16* Ai, int ldai, const bf16* Wi, int ldwi, int Ki,
                   const bf16* h_cur, const bf16* Wh, bf16* h_new,
                   int m0, int n0, int lane) {
  v8f gia[3] = {vz(), vz(), vz()};
  v8f gha[3] = {vz(), vz(), vz()};
  if (Ai) accum_gates(gia, Ai, ldai, Wi, ldwi, Ki, H, m0, n0, lane);
  accum_gates(gha, h_cur, H, Wh, H, H, H, m0, n0, lane);
  int li = lane & 15, hb = lane >> 4, col = n0 + li;
#pragma unroll
  for (int i = 0; i < 8; ++i) {
    int row = m0 + i + 8 * hb;
    size_t gb = (size_t)row * (3 * H);
    float gir = gia[0][i], giz = gia[1][i], ginn = gia[2][i];
    if (gi_t) { gir += (float)gi_t[gb + col]; giz += (float)gi_t[gb + col + H];
                ginn += (float)gi_t[gb + col + 2 * H]; }
    if (bi)   { gir += bi[col]; giz += bi[col + H]; ginn += bi[col + 2 * H]; }
    float ghr = gha[0][i] + bh[col];
    float ghz = gha[1][i] + bh[col + H];
    float ghn = gha[2][i] + bh[col + 2 * H];
    float r = sigmoidf_(gir + ghr);
    float z = sigmoidf_(giz + ghz);
    float nn = tanhf(ginn + r * ghn);
    float ho = (float)h_cur[(size_t)row * H + col];
    float h = (1.f - z) * nn + z * ho;
    h = fminf(fmaxf(h, -CLIPV), CLIPV);
    h_new[(size_t)row * H + col] = (bf16)h;
  }
}

// u = mean + eps * exp(0.5*logvar); both heads computed in one wave.
DEV void sample_phase(const bf16* A, int K, const bf16* Wm, const float* bm,
                      const bf16* Wv, const float* bv, int N, const float* eps_t,
                      float* uoutf, bf16* uoutbf, int ubld, int ubcol0,
                      int m0, int n0, int lane) {
  v8f am = vz(), av = vz();
  for (int k0 = 0; k0 < K; k0 += 32) {
    v16bf a = load_a_frag(A, K, m0, k0, lane);
    am = wmma_bf(a, load_b_frag(Wm, K, n0, k0, lane), am);
    av = wmma_bf(a, load_b_frag(Wv, K, n0, k0, lane), av);
  }
  int li = lane & 15, hb = lane >> 4, col = n0 + li;
#pragma unroll
  for (int i = 0; i < 8; ++i) {
    int row = m0 + i + 8 * hb;
    float um = am[i] + bm[col];
    float lv = av[i] + bv[col];
    float u = um + eps_t[(size_t)row * N + col] * __expf(0.5f * lv);
    if (uoutf) uoutf[(size_t)row * N + col] = u;
    uoutbf[(size_t)row * ubld + ubcol0 + col] = (bf16)u;
  }
}

DEV void spike_obs(const DecArgs& g, int m0, int n0, int lane, int t_main) {
  v8f acc = vz();
  for (int k0 = 0; k0 < 192; k0 += 32)
    acc = wmma_bf(load_a_frag(g.gin, 192, m0, k0, lane),
                  load_b_frag(g.Ws, 192, n0, k0, lane), acc);
  int li = lane & 15, hb = lane >> 4, col = n0 + li;
  float bsv = g.b_s[col];
#pragma unroll
  for (int i = 0; i < 8; ++i) {
    int row = m0 + i + 8 * hb;
    float pre = acc[i] + bsv;
    float spike = fmaxf(__expf(pre) - 1.f, 0.f);
    size_t oidx = (size_t)row * N_ + col;
    float no = spike + DECAY * g.obs_f32[oidx];   // BIAS=0, GAIN=1
    g.obs_f32[oidx] = no;
    g.obs_bf[oidx] = (bf16)no;
    if (t_main >= 0) {
      g.out_spikes[((size_t)t_main * B_ + row) * N_ + col] = spike;
      g.out_data[((size_t)row * T_ + t_main) * N_ + col] = no;
    }
  }
}

__global__ void __launch_bounds__(256) decode_kernel(DecArgs g) {
  unsigned ph = 0;
  int lane = threadIdx.x & 31;
  int w = (int)((blockIdx.x * blockDim.x + threadIdx.x) >> 5);   // 0..255

  // warmup obs steps: factor part of gin is zero (pre-zeroed buffer)
  for (int s = 0; s < PAD_; ++s) {
    const bf16* oc = (s & 1) ? g.octrl1 : g.octrl0;
    bf16* on = (s & 1) ? g.octrl0 : g.octrl1;
    if (w < 128)
      gru_phase(g.Gi_o + (size_t)s * B_ * 768, nullptr, OC, g.bh_o,
                g.obs_bf, 512, g.Wio + 512, 1024, 512,
                oc, g.Who, on, (w >> 4) * 16, (w & 15) * 16, lane);
    gbar(g.counter, g.nblocks, ph);
    if (w < 64)
      sample_phase(on, OC, g.Wom, g.b_om, g.Wolv, g.b_olv, OL,
                   g.eps_obs + (size_t)s * B_ * OL, nullptr, g.gin, 192, 0,
                   (w >> 3) * 16, (w & 7) * 16, lane);
    gbar(g.counter, g.nblocks, ph);
    spike_obs(g, (w >> 5) * 16, (w & 31) * 16, lane, -1);
    gbar(g.counter, g.nblocks, ph);
  }

  for (int t = 0; t < T_; ++t) {
    const bf16* dc = (t & 1) ? g.dctrl1 : g.dctrl0;
    bf16* dn = (t & 1) ? g.dctrl0 : g.dctrl1;
    const bf16* gc = (t & 1) ? g.gen1 : g.gen0;
    bf16* gn = (t & 1) ? g.gen0 : g.gen1;
    int os = t + PAD_;
    const bf16* oc = (os & 1) ? g.octrl1 : g.octrl0;
    bf16* on = (os & 1) ? g.octrl0 : g.octrl1;

    // A: d_ctrl update (H=128); gi = Gi_c[t] (incl bi) + factor@Wi[:,512:]^T
    if (w < 64)
      gru_phase(g.Gi_c + (size_t)t * B_ * 384, nullptr, DC, g.bh_d,
                g.factor_bf, F_, g.Wid + 512, 576, F_,
                dc, g.Whd, dn, (w >> 3) * 16, (w & 7) * 16, lane);
    gbar(g.counter, g.nblocks, ph);

    // B: deep u sample (N=16), write gen_inputs output + padded bf16 u
    if (w < 8)
      sample_phase(dn, DC, g.Wum, g.b_um, g.Wulv, g.b_ulv, UL,
                   g.eps_deep + (size_t)t * B_ * UL,
                   g.out_geninputs + (size_t)t * B_ * UL,
                   g.u_bf, 32, 0, w * 16, 0, lane);
    gbar(g.counter, g.nblocks, ph);

    // C: d_gen update (H=256); gi = u@Wig^T + bi_g  (Wig zero-padded to K=32)
    if (w < 128)
      gru_phase(nullptr, g.bi_g, GEN, g.bh_g,
                g.u_bf, 32, g.Wig, 32, 32,
                gc, g.Whg, gn, (w >> 4) * 16, (w & 15) * 16, lane);
    gbar(g.counter, g.nblocks, ph);

    // D: factor = gen @ Wf^T (N=64); feed outputs + gin[:,128:192]
    if (w < 32) {
      int m0 = (w >> 2) * 16, n0 = (w & 3) * 16;
      v8f acc = vz();
      for (int k0 = 0; k0 < GEN; k0 += 32)
        acc = wmma_bf(load_a_frag(gn, GEN, m0, k0, lane),
                      load_b_frag(g.Wf, GEN, n0, k0, lane), acc);
      int li = lane & 15, hb = lane >> 4, col = n0 + li;
#pragma unroll
      for (int i = 0; i < 8; ++i) {
        int row = m0 + i + 8 * hb;
        float f = acc[i];
        g.out_factors[((size_t)t * B_ + row) * F_ + col] = f;
        bf16 fb = (bf16)f;
        g.factor_bf[(size_t)row * F_ + col] = fb;
        g.factors_all[((size_t)t * B_ + row) * F_ + col] = fb;
        g.gin[(size_t)row * 192 + 128 + col] = fb;
      }
    }
    gbar(g.counter, g.nblocks, ph);

    // E: o_ctrl update (H=256); gi = Gi_o[os] + obs_state@Wio[:,512:]^T
    if (w < 128)
      gru_phase(g.Gi_o + (size_t)os * B_ * 768, nullptr, OC, g.bh_o,
                g.obs_bf, 512, g.Wio + 512, 1024, 512,
                oc, g.Who, on, (w >> 4) * 16, (w & 15) * 16, lane);
    gbar(g.counter, g.nblocks, ph);

    // F: obs u sample (N=128) -> gin[:, :128]
    if (w < 64)
      sample_phase(on, OC, g.Wom, g.b_om, g.Wolv, g.b_olv, OL,
                   g.eps_obs + (size_t)os * B_ * OL, nullptr, g.gin, 192, 0,
                   (w >> 3) * 16, (w & 7) * 16, lane);
    gbar(g.counter, g.nblocks, ph);

    // G: spike + obs_state + outputs
    spike_obs(g, (w >> 5) * 16, (w & 31) * 16, lane, t);
    gbar(g.counter, g.nblocks, ph);
  }
}

// ---------------------------------------------------------------------------
// Host driver
// ---------------------------------------------------------------------------
extern "C" void kernel_launch(void* const* d_in, const int* in_sizes, int n_in,
                              void* d_out, int out_size, void* d_ws, size_t ws_size,
                              hipStream_t stream) {
  (void)in_sizes; (void)n_in; (void)out_size; (void)ws_size;
  const float* x        = (const float*)d_in[0];
  const float* eps_g0   = (const float*)d_in[1];
  const float* eps_deep = (const float*)d_in[2];
  const float* eps_obs  = (const float*)d_in[3];
  const float *enc_Wi[6], *enc_Wh[6], *enc_bi[6], *enc_bh[6];
  for (int e = 0; e < 6; ++e) {
    enc_Wi[e] = (const float*)d_in[4 + 4 * e];
    enc_Wh[e] = (const float*)d_in[5 + 4 * e];
    enc_bi[e] = (const float*)d_in[6 + 4 * e];
    enc_bh[e] = (const float*)d_in[7 + 4 * e];
  }
  const float* W_gm  = (const float*)d_in[28]; const float* b_gm  = (const float*)d_in[29];
  const float* W_glv = (const float*)d_in[30]; const float* b_glv = (const float*)d_in[31];
  const float* W_gs  = (const float*)d_in[32]; const float* b_gs  = (const float*)d_in[33];
  const float* Wi_d  = (const float*)d_in[34]; const float* Wh_d  = (const float*)d_in[35];
  const float* bi_d  = (const float*)d_in[36]; const float* bh_d  = (const float*)d_in[37];
  const float* W_um  = (const float*)d_in[38]; const float* b_um  = (const float*)d_in[39];
  const float* W_ulv = (const float*)d_in[40]; const float* b_ulv = (const float*)d_in[41];
  const float* Wi_g  = (const float*)d_in[42]; const float* Wh_g  = (const float*)d_in[43];
  const float* bi_g  = (const float*)d_in[44]; const float* bh_g  = (const float*)d_in[45];
  const float* W_f   = (const float*)d_in[46];
  const float* Wi_o  = (const float*)d_in[47]; const float* Wh_o  = (const float*)d_in[48];
  const float* bi_o  = (const float*)d_in[49]; const float* bh_o  = (const float*)d_in[50];
  const float* W_om  = (const float*)d_in[51]; const float* b_om  = (const float*)d_in[52];
  const float* W_olv = (const float*)d_in[53]; const float* b_olv = (const float*)d_in[54];
  const float* W_s   = (const float*)d_in[55]; const float* b_s   = (const float*)d_in[56];
  const float* W_lr  = (const float*)d_in[57]; const float* b_lr  = (const float*)d_in[58];

  float* out_rates     = (float*)d_out;
  float* out_data      = out_rates + (size_t)T_ * B_ * N_;
  float* out_spikes    = out_data + (size_t)B_ * T_ * N_;
  float* out_factors   = out_spikes + (size_t)T_ * B_ * N_;
  float* out_geninputs = out_factors + (size_t)T_ * B_ * F_;

  // ---- workspace layout ----
  char* base = (char*)d_ws;
  size_t off = 0;
  auto take = [&](size_t bytes) -> char* {
    char* p = base + off;
    off = (off + bytes + 255) & ~(size_t)255;
    return p;
  };
  // zero block (memset each call)
  unsigned* counter = (unsigned*)take(256);
  bf16* dctrl0 = (bf16*)take((size_t)B_ * DC * 2);
  bf16* dctrl1 = (bf16*)take((size_t)B_ * DC * 2);
  bf16* octrl0 = (bf16*)take((size_t)B_ * OC * 2);
  bf16* octrl1 = (bf16*)take((size_t)B_ * OC * 2);
  float* obs_f32 = (float*)take((size_t)B_ * N_ * 4);
  bf16* obs_bf = (bf16*)take((size_t)B_ * N_ * 2);
  bf16* u_bf   = (bf16*)take((size_t)B_ * 32 * 2);
  bf16* gin    = (bf16*)take((size_t)B_ * 192 * 2);
  size_t zero_bytes = off;
  // rest
  bf16* gen0 = (bf16*)take((size_t)B_ * GEN * 2);
  bf16* gen1 = (bf16*)take((size_t)B_ * GEN * 2);
  bf16* factor_bf = (bf16*)take((size_t)B_ * F_ * 2);
  float* gmean = (float*)take((size_t)B_ * GL * 4);
  float* glv   = (float*)take((size_t)B_ * GL * 4);
  float* g0    = (float*)take((size_t)B_ * GL * 4);
  float* genstate_f32 = (float*)take((size_t)B_ * GEN * 4);
  float* genc_f32     = (float*)take((size_t)B_ * 512 * 4);
  bf16* WiBf[6]; bf16* WhBf[6];
  for (int e = 0; e < 6; ++e) {
    WiBf[e] = (bf16*)take((size_t)768 * 512 * 2);
    WhBf[e] = (bf16*)take((size_t)768 * 256 * 2);
  }
  bf16* WidBf  = (bf16*)take((size_t)384 * 576 * 2);
  bf16* WhdBf  = (bf16*)take((size_t)384 * 128 * 2);
  bf16* WumBf  = (bf16*)take((size_t)16 * 128 * 2);
  bf16* WulvBf = (bf16*)take((size_t)16 * 128 * 2);
  bf16* WigBf  = (bf16*)take((size_t)768 * 32 * 2);     // K padded 16 -> 32
  bf16* WhgBf  = (bf16*)take((size_t)768 * 256 * 2);
  bf16* WfBf   = (bf16*)take((size_t)64 * 256 * 2);
  bf16* WioBf  = (bf16*)take((size_t)768 * 1024 * 2);
  bf16* WhoBf  = (bf16*)take((size_t)768 * 256 * 2);
  bf16* WomBf  = (bf16*)take((size_t)128 * 256 * 2);
  bf16* WolvBf = (bf16*)take((size_t)128 * 256 * 2);
  bf16* WsBf   = (bf16*)take((size_t)512 * 192 * 2);
  bf16* WlrBf  = (bf16*)take((size_t)512 * 64 * 2);
  size_t xpad_bytes = (size_t)TP * B_ * N_ * 2;
  bf16* x_pad = (bf16*)take(xpad_bytes);
  bf16* GiEnc[6];
  for (int e = 0; e < 6; ++e)
    GiEnc[e] = (bf16*)take((size_t)(e < 2 ? TP : T_) * B_ * 768 * 2);
  bf16* out_obs = (bf16*)take((size_t)TP * B_ * 512 * 2);
  bf16* out_c   = (bf16*)take((size_t)T_ * B_ * 512 * 2);
  bf16* Gi_c    = (bf16*)take((size_t)T_ * B_ * 384 * 2);
  bf16* Gi_o    = (bf16*)take((size_t)TP * B_ * 768 * 2);
  bf16* factors_all = (bf16*)take((size_t)T_ * B_ * F_ * 2);

  auto cdiv = [](long n, long d) { return (int)((n + d - 1) / d); };

  hipMemsetAsync(d_ws, 0, zero_bytes, stream);
  hipMemsetAsync(x_pad, 0, xpad_bytes, stream);

  // weight casts fp32 -> bf16 (zero-fills when dst_stride > src cols)
  auto cast = [&](bf16* dst, const float* src, int rows, int scols, int dstride) {
    int n = rows * dstride;
    cast_rows_kernel<<<cdiv(n, 256), 256, 0, stream>>>(dst, src, rows, scols, dstride);
  };
  for (int e = 0; e < 6; ++e) {
    cast(WiBf[e], enc_Wi[e], 768, 512, 512);
    cast(WhBf[e], enc_Wh[e], 768, 256, 256);
  }
  cast(WidBf, Wi_d, 384, 576, 576);   cast(WhdBf, Wh_d, 384, 128, 128);
  cast(WumBf, W_um, 16, 128, 128);    cast(WulvBf, W_ulv, 16, 128, 128);
  cast(WigBf, Wi_g, 768, 16, 32);     cast(WhgBf, Wh_g, 768, 256, 256);
  cast(WfBf, W_f, 64, 256, 256);
  cast(WioBf, Wi_o, 768, 1024, 1024); cast(WhoBf, Wh_o, 768, 256, 256);
  cast(WomBf, W_om, 128, 256, 256);   cast(WolvBf, W_olv, 128, 256, 256);
  cast(WsBf, W_s, 512, 192, 192);     cast(WlrBf, W_lr, 512, 64, 64);

  transpose_cast_x_kernel<<<cdiv((long)T_ * B_ * N_, 256), 256, 0, stream>>>(x_pad, x);

  // input projections for all encoders (big WMMA GEMMs)
  for (int e = 0; e < 6; ++e) {
    const bf16* A = (e < 2) ? x_pad : (x_pad + (size_t)PAD_ * B_ * N_);
    int M = (e < 2 ? TP : T_) * B_;
    gemm_bf16_kernel<<<1024, 256, 0, stream>>>(A, 512, WiBf[e], 512, enc_bi[e],
                                               GiEnc[e], nullptr, M, 768, 512, 0);
  }

  // persistent bi-GRU encoders (6 directions in one launch)
  EncArgs ea;
  ea.e[0] = {GiEnc[0], WhBf[0], enc_bh[0], out_obs, nullptr, TP, 0};
  ea.e[1] = {GiEnc[1], WhBf[1], enc_bh[1], out_obs, nullptr, TP, 1};
  ea.e[2] = {GiEnc[2], WhBf[2], enc_bh[2], nullptr, genc_f32, T_, 0};
  ea.e[3] = {GiEnc[3], WhBf[3], enc_bh[3], nullptr, genc_f32, T_, 1};
  ea.e[4] = {GiEnc[4], WhBf[4], enc_bh[4], out_c, nullptr, T_, 0};
  ea.e[5] = {GiEnc[5], WhBf[5], enc_bh[5], out_c, nullptr, T_, 1};
  encoder_kernel<<<6, 1024, (size_t)B_ * HSTR * sizeof(bf16), stream>>>(ea);

  // g path (tiny)
  lin_naive_kernel<<<cdiv(B_ * GL, 256), 256, 0, stream>>>(gmean, nullptr, genc_f32,
                                                           W_gm, b_gm, B_, GL, 512);
  lin_naive_kernel<<<cdiv(B_ * GL, 256), 256, 0, stream>>>(glv, nullptr, genc_f32,
                                                           W_glv, b_glv, B_, GL, 512);
  sample_g0_kernel<<<cdiv(B_ * GL, 256), 256, 0, stream>>>(g0, gmean, glv, eps_g0,
                                                           B_ * GL);
  lin_naive_kernel<<<cdiv(B_ * GEN, 256), 256, 0, stream>>>(genstate_f32, gen0, g0,
                                                            W_gs, b_gs, B_, GEN, GL);
  lin_naive_kernel<<<cdiv(B_ * F_, 256), 256, 0, stream>>>(nullptr, factor_bf,
                                                           genstate_f32, W_f, nullptr,
                                                           B_, F_, GEN);

  // decoder input projections (time-parallel parts)
  gemm_bf16_kernel<<<1024, 256, 0, stream>>>(out_c, 512, WidBf, 576, bi_d,
                                             Gi_c, nullptr, T_ * B_, 384, 512, 0);
  gemm_bf16_kernel<<<1024, 256, 0, stream>>>(out_obs, 512, WioBf, 1024, bi_o,
                                             Gi_o, nullptr, TP * B_, 768, 512, 0);

  // persistent decoder
  DecArgs da;
  da.counter = counter; da.nblocks = 32;
  da.Wid = WidBf; da.Whd = WhdBf; da.Wum = WumBf; da.Wulv = WulvBf;
  da.Wig = WigBf; da.Whg = WhgBf; da.Wf = WfBf; da.Wio = WioBf; da.Who = WhoBf;
  da.Wom = WomBf; da.Wolv = WolvBf; da.Ws = WsBf;
  da.bh_d = bh_d; da.b_um = b_um; da.b_ulv = b_ulv; da.bi_g = bi_g; da.bh_g = bh_g;
  da.bh_o = bh_o; da.b_om = b_om; da.b_olv = b_olv; da.b_s = b_s;
  da.Gi_c = Gi_c; da.Gi_o = Gi_o;
  da.eps_deep = eps_deep; da.eps_obs = eps_obs;
  da.dctrl0 = dctrl0; da.dctrl1 = dctrl1; da.gen0 = gen0; da.gen1 = gen1;
  da.octrl0 = octrl0; da.octrl1 = octrl1;
  da.factor_bf = factor_bf; da.obs_bf = obs_bf; da.u_bf = u_bf; da.gin = gin;
  da.factors_all = factors_all; da.obs_f32 = obs_f32;
  da.out_factors = out_factors; da.out_geninputs = out_geninputs;
  da.out_spikes = out_spikes; da.out_data = out_data;
  decode_kernel<<<32, 256, 0, stream>>>(da);

  // rates = exp(factors @ Wlr^T + b)
  gemm_bf16_kernel<<<1024, 256, 0, stream>>>(factors_all, 64, WlrBf, 64, b_lr,
                                             nullptr, out_rates, T_ * B_, 512, 64, 1);
}
